// DefConv_5428838662193
// MI455X (gfx1250) — matrix operand order repaired and checked by reading, not compiled, not run on hardware
//
#include <hip/hip_runtime.h>
#include <hip/hip_bf16.h>

// ---------------------------------------------------------------------------
// DCNv2 (modulated deformable conv) + BN + ReLU for gfx1250 (MI455X).
// ~23 GFLOP vs ~110MB traffic (~5us at 23.3TB/s) => matrix-op bound.
// Both GEMMs use v_wmma_f32_16x16x32_bf16 (f32 accum). Main GEMM feeds its
// LDS tiles exclusively via the Tensor Data Mover (tensor_load_to_lds,
// TENSORcnt double-buffered), with TDM LDS padding producing the
// bank-conflict-free 40-half row stride directly.
// ---------------------------------------------------------------------------

#define C_IN   256
#define C_OUT  256
#define K2     9
#define DG     2
#define CG     128            // C_IN / DG
#define H      64
#define W      64
#define NB     4
#define NPIX   4096           // H*W
#define KOFF   (K2 * C_IN)    // 2304 : offset-conv GEMM K (tap-major, then channel)
#define KMAIN  (DG * CG * K2) // 2304 : main GEMM K (group, tap, channel)
#define OC_OFF 54             // 3*DG*K2
#define OC_PAD 64
#define LDSTR  40             // LDS row stride in halves (80B: 16B aligned, conflict-free)
#define EPSBN  1e-5f

typedef __bf16 bf16;
typedef __attribute__((ext_vector_type(16))) __bf16 v16bf;
typedef __attribute__((ext_vector_type(8)))  __bf16 v8bf;
typedef __attribute__((ext_vector_type(8)))  float  v8f;
typedef __attribute__((ext_vector_type(4)))  unsigned int u32x4;
typedef __attribute__((ext_vector_type(8)))  int i32x8;
typedef __attribute__((ext_vector_type(4)))  int i32x4;

// ---- fragment loaders per CDNA5 ISA 7.12.2 --------------------------------
// A (16x32 bf16): lane L holds M=L%16; K = (L/16)*8 + {0..7} and +16..+23.
__device__ __forceinline__ v16bf frag_a(const bf16* lds, int row, int half) {
  const bf16* p = lds + row * LDSTR + half * 8;
  v8bf lo = *(const v8bf*)(p);
  v8bf hi = *(const v8bf*)(p + 16);
  return __builtin_shufflevector(lo, hi, 0,1,2,3,4,5,6,7,8,9,10,11,12,13,14,15);
}
// B (32x16 bf16): lane L holds N=L%16; K = (L/16)*8 + {0..7} and +16..+23.
__device__ __forceinline__ v16bf frag_b(const bf16* lds, int col, int half) {
  const bf16* p = lds + col * LDSTR + half * 8;
  v8bf lo = *(const v8bf*)(p);
  v8bf hi = *(const v8bf*)(p + 16);
  return __builtin_shufflevector(lo, hi, 0,1,2,3,4,5,6,7,8,9,10,11,12,13,14,15);
}

// ---- Tensor Data Mover: 2D tile [128 rows x 32 bf16], row stride KMAIN ----
// D# per CDNA5 ISA:
//  g0: count=1 | lds_addr | global_addr[56:0] | type=2
//  g1: data_size=1(2B), pad_enable, pad_interval=3 (16 DW = 64B),
//      pad_amount=3 (4 DW = 16B) => LDS row stride 80B == LDSTR halves.
//      tensor_dim0=KMAIN, tensor_dim1=big, tile_dim0=32, tile_dim1=128,
//      tensor_dim0_stride=KMAIN.
// LDS flat pointer low 32 bits are the LDS byte offset (aperture truncation).
// This toolchain's builtin is the 6-arg form:
//   (u32x4 g0, i32x8 g1, i32x4, i32x4, i32x8, i32 cpol)
__device__ __forceinline__ void tdm_load_tile_128x32(const bf16* gsrc, void* ldst) {
  unsigned long long ga = (unsigned long long)(uintptr_t)gsrc;
  unsigned lds_addr = (unsigned)(uintptr_t)ldst;
  u32x4 g0;
  g0[0] = 1u;                                        // count=1, user descriptor
  g0[1] = lds_addr;                                  // lds_addr [63:32]
  g0[2] = (unsigned)(ga & 0xffffffffu);              // global_addr lo
  g0[3] = (unsigned)((ga >> 32) & 0x01ffffffu) | (2u << 30);  // addr hi | type=2
  i32x8 g1;
  g1[0] = (1 << 16) | (1 << 20) | (3 << 22) | (3 << 25); // 2B, pad en, 16DW/+4DW
  g1[1] = (KMAIN & 0xffff) << 16;                    // tensor_dim0[15:0] @ [63:48]
  g1[2] = (KMAIN >> 16) | (0 << 16);                 // dim0[31:16] | tensor_dim1[15:0]
  g1[3] = 0x4000 | (32 << 16);                       // dim1[31:16]=0x4000 | tile_dim0=32
  g1[4] = 128;                                       // tile_dim1=128, tile_dim2=0
  g1[5] = KMAIN;                                     // tensor_dim0_stride lo32
  g1[6] = 0;                                         // stride hi16 | dim1_stride lo16
  g1[7] = 0;
  i32x4 z4 = {0, 0, 0, 0};
  i32x8 z8 = {0, 0, 0, 0, 0, 0, 0, 0};
  __builtin_amdgcn_tensor_load_to_lds(g0, g1, z4, z4, z8, 0);
}

// ---------------------------------------------------------------------------
// Kernel 0: weight reorder + fp32->bf16.
//   w_bf [C_OUT][KMAIN],  k = g*1152 + tap*128 + cg
//   wof_bf [64 pad][KOFF], k = tap*256 + c
// ---------------------------------------------------------------------------
__global__ void __launch_bounds__(256) prep_weights(
    const float* __restrict__ w, const float* __restrict__ w_off,
    bf16* __restrict__ w_bf, bf16* __restrict__ wof_bf) {
  int idx = blockIdx.x * 256 + threadIdx.x;
  const int NW = C_OUT * KMAIN;
  if (idx < NW) {
    int o = idx / KMAIN, kk = idx % KMAIN;
    int g = kk / (CG * K2), rem = kk % (CG * K2);
    int tap = rem / CG, cg = rem % CG;
    int c = g * CG + cg;
    w_bf[idx] = (bf16)w[(o * C_IN + c) * K2 + tap];
  } else {
    int j = idx - NW;
    if (j < OC_PAD * KOFF) {
      int oc = j / KOFF, kk = j % KOFF;
      int tap = kk / C_IN, c = kk % C_IN;
      float v = (oc < OC_OFF) ? w_off[(oc * C_IN + c) * K2 + tap] : 0.f;
      wof_bf[j] = (bf16)v;
    }
  }
}

// ---------------------------------------------------------------------------
// Kernel 1: offset/mask conv as WMMA GEMM. M=64(pad of 54), K=2304, N=4096/batch.
// Tile M=64 x N=256, 8 waves (2x4), wave = 32(M) x 64(N). B tile is on-the-fly
// zero-padded im2col of x (one tap per 32-wide k-step).
// ---------------------------------------------------------------------------
__global__ void __launch_bounds__(256) offset_conv_wmma(
    const float* __restrict__ x, const bf16* __restrict__ wof,
    const float* __restrict__ b_off, float* __restrict__ om) {
  __shared__ __align__(16) bf16 Alds[OC_PAD * LDSTR];
  __shared__ __align__(16) bf16 Blds[256 * LDSTR];
  const int tid  = threadIdx.x;
  const int lane = tid & 31, wv = tid >> 5;
  const int half = lane >> 4, l16 = lane & 15;
  const int wm = wv & 1, wn = wv >> 1;          // 2 x 4 wave grid
  const int b  = blockIdx.y;
  const int n0 = blockIdx.x * 256;
  const int gn = n0 + tid;                      // pixel this thread stages
  const int py = gn >> 6, px = gn & 63;
  const float* xb = x + (size_t)b * C_IN * NPIX;

  v8f acc[2][4];
  #pragma unroll
  for (int i = 0; i < 2; i++)
    #pragma unroll
    for (int j = 0; j < 4; j++)
      #pragma unroll
      for (int e = 0; e < 8; e++) acc[i][j][e] = 0.f;

  for (int k0 = 0; k0 < KOFF; k0 += 32) {
    { // A tile 64x32: 256 threads x one b128
      int row = tid >> 2, seg = tid & 3;
      *(v8bf*)(Alds + row * LDSTR + seg * 8) =
          *(const v8bf*)(wof + (size_t)row * KOFF + k0 + seg * 8);
    }
    { // B tile: 32 channels of x at tap (k0/256), zero-padded borders
      int t  = k0 >> 8, c0 = k0 & 255;
      int dy = t / 3 - 1, dx = t % 3 - 1;
      int yy = py + dy, xc = px + dx;
      bool ok = (yy >= 0) && (yy < H) && (xc >= 0) && (xc < W);
      const float* xp = xb + (size_t)c0 * NPIX + yy * W + xc;
      #pragma unroll 4
      for (int i = 0; i < 32; i++) {
        float v = ok ? xp[i * NPIX] : 0.f;
        Blds[tid * LDSTR + i] = (bf16)v;
      }
    }
    __syncthreads();
    v16bf afr[2], bfr[4];
    #pragma unroll
    for (int mi = 0; mi < 2; mi++) afr[mi] = frag_a(Alds, wm * 32 + mi * 16 + l16, half);
    #pragma unroll
    for (int ni = 0; ni < 4; ni++) bfr[ni] = frag_b(Blds, wn * 64 + ni * 16 + l16, half);
    #pragma unroll
    for (int mi = 0; mi < 2; mi++)
      #pragma unroll
      for (int ni = 0; ni < 4; ni++)
        acc[mi][ni] = __builtin_amdgcn_wmma_f32_16x16x32_bf16(
            false, afr[mi], false, bfr[ni], (short)0, acc[mi][ni], false, false);
    __syncthreads();
  }
  #pragma unroll
  for (int mi = 0; mi < 2; mi++) {
    #pragma unroll
    for (int r = 0; r < 8; r++) {
      int oc = wm * 32 + mi * 16 + half * 8 + r;
      if (oc < OC_OFF) {
        float bias = b_off[oc];
        #pragma unroll
        for (int ni = 0; ni < 4; ni++) {
          int n = n0 + wn * 64 + ni * 16 + l16;
          om[((size_t)b * OC_OFF + oc) * NPIX + n] = acc[mi][ni][r] + bias;
        }
      }
    }
  }
}

// ---------------------------------------------------------------------------
// Kernel 2: bilinear sampling + modulation -> TRANSPOSED bf16 val matrix:
// valt[b][n][kfull], kfull = g*1152 + tap*128 + c. Each thread owns one pixel
// and writes 128 consecutive channels (16 x nontemporal b128 stores), so the
// main GEMM B tile is a plain strided 2D tile -> TDM-loadable, no transpose.
// ---------------------------------------------------------------------------
__global__ void __launch_bounds__(256) deform_sample(
    const float* __restrict__ x, const float* __restrict__ om,
    bf16* __restrict__ valt) {
  const int b  = blockIdx.z;
  const int gt = blockIdx.y;                    // g*9 + tap
  const int g = gt / K2, tap = gt % K2;
  const int n  = blockIdx.x * 256 + threadIdx.x;
  const int yy = n >> 6, xx = n & 63;
  const float* omb = om + (size_t)b * OC_OFF * NPIX;
  float offy = omb[(size_t)gt * NPIX + n];
  float offx = omb[(size_t)(18 + gt) * NPIX + n];
  float mraw = omb[(size_t)(36 + gt) * NPIX + n];
  float mask = 1.f / (1.f + __expf(-mraw));
  float fy = offy + (float)(tap / 3 - 1) + (float)yy;
  float fx = offx + (float)(tap % 3 - 1) + (float)xx;
  float y0f = floorf(fy), x0f = floorf(fx);
  float ly = fy - y0f, lx = fx - x0f;
  int y0 = (int)y0f, x0 = (int)x0f, y1 = y0 + 1, x1 = x0 + 1;
  bool vy0 = (y0 >= 0) && (y0 < H), vy1 = (y1 >= 0) && (y1 < H);
  bool vx0 = (x0 >= 0) && (x0 < W), vx1 = (x1 >= 0) && (x1 < W);
  int cy0 = min(max(y0, 0), H - 1), cy1 = min(max(y1, 0), H - 1);
  int cx0 = min(max(x0, 0), W - 1), cx1 = min(max(x1, 0), W - 1);
  int i00 = cy0 * W + cx0, i01 = cy0 * W + cx1;
  int i10 = cy1 * W + cx0, i11 = cy1 * W + cx1;
  float w00 = (1.f - ly) * (1.f - lx) * ((vy0 && vx0) ? 1.f : 0.f) * mask;
  float w01 = (1.f - ly) * lx         * ((vy0 && vx1) ? 1.f : 0.f) * mask;
  float w10 = ly * (1.f - lx)         * ((vy1 && vx0) ? 1.f : 0.f) * mask;
  float w11 = ly * lx                 * ((vy1 && vx1) ? 1.f : 0.f) * mask;
  const float* xp = x + ((size_t)b * C_IN + g * CG) * NPIX;
  bf16* vout = valt + ((size_t)b * NPIX + n) * KMAIN + g * (CG * K2) + tap * CG;
  #pragma unroll 4
  for (int c8 = 0; c8 < CG; c8 += 8) {
    v8bf pack;
    #pragma unroll
    for (int j = 0; j < 8; j++) {
      const float* xc = xp + (size_t)(c8 + j) * NPIX;
      float v = w00 * xc[i00] + w01 * xc[i01] + w10 * xc[i10] + w11 * xc[i11];
      pack[j] = (bf16)v;
    }
    __builtin_nontemporal_store(pack, (v8bf*)(vout + c8));
  }
}

// ---------------------------------------------------------------------------
// Kernel 3: main GEMM (M=256, K=2304, N=4096/batch) + fused BN + ReLU.
// 8 waves (4 along M x 2 along N), tile 128(M) x 128(N). Both LDS tiles are
// 128x32 bf16 strided tiles DMA'd by the Tensor Data Mover (wave 0 issues,
// double-buffered on TENSORcnt). TDM pad config yields the 40-half row stride
// the b128 fragment reads expect.
// ---------------------------------------------------------------------------
__global__ void __launch_bounds__(256) main_gemm_wmma(
    const bf16* __restrict__ wbf, const bf16* __restrict__ valt,
    const float* __restrict__ bias, const float* __restrict__ gamma,
    const float* __restrict__ beta, const float* __restrict__ mean,
    const float* __restrict__ var, float* __restrict__ out) {
  __shared__ __align__(16) bf16 Alds[2][128 * LDSTR];
  __shared__ __align__(16) bf16 Blds[2][128 * LDSTR];
  const int tid  = threadIdx.x;
  const int lane = tid & 31, wv = tid >> 5;
  const int half = lane >> 4, l16 = lane & 15;
  const int wm = wv & 3, wn = wv >> 2;          // 4 x 2 wave grid
  const int b  = blockIdx.z;
  const int m0 = blockIdx.y * 128;
  const int n0 = blockIdx.x * 128;
  const bf16* aG = wbf + (size_t)m0 * KMAIN;                  // rows m0..m0+127
  const bf16* bG = valt + ((size_t)b * NPIX + n0) * KMAIN;    // rows n0..n0+127

  v8f acc[2][4];
  #pragma unroll
  for (int i = 0; i < 2; i++)
    #pragma unroll
    for (int j = 0; j < 4; j++)
      #pragma unroll
      for (int e = 0; e < 8; e++) acc[i][j][e] = 0.f;

  if (wv == 0) {                                // prologue DMA into buffer 0
    tdm_load_tile_128x32(aG, Alds[0]);
    tdm_load_tile_128x32(bG, Blds[0]);
  }
  int p = 0;
  for (int k0 = 0; k0 < KMAIN; k0 += 32) {
    if (wv == 0) {
      if (k0 + 32 < KMAIN) {                    // prefetch next pair, then wait
        tdm_load_tile_128x32(aG + k0 + 32, Alds[p ^ 1]);
        tdm_load_tile_128x32(bG + k0 + 32, Blds[p ^ 1]);
        __builtin_amdgcn_s_wait_tensorcnt(2);   // current pair complete
      } else {
        __builtin_amdgcn_s_wait_tensorcnt(0);
      }
    }
    __syncthreads();
    v16bf afr[2], bfr[4];
    #pragma unroll
    for (int mi = 0; mi < 2; mi++) afr[mi] = frag_a(Alds[p], wm * 32 + mi * 16 + l16, half);
    #pragma unroll
    for (int ni = 0; ni < 4; ni++) bfr[ni] = frag_b(Blds[p], wn * 64 + ni * 16 + l16, half);
    #pragma unroll
    for (int mi = 0; mi < 2; mi++)
      #pragma unroll
      for (int ni = 0; ni < 4; ni++)
        acc[mi][ni] = __builtin_amdgcn_wmma_f32_16x16x32_bf16(
            false, afr[mi], false, bfr[ni], (short)0, acc[mi][ni], false, false);
    __syncthreads();                            // compute done before DMA reuses buf
    p ^= 1;
  }
  // fused (conv_bias -> BN eval -> ReLU)
  #pragma unroll
  for (int mi = 0; mi < 2; mi++) {
    #pragma unroll
    for (int r = 0; r < 8; r++) {
      int o = m0 + wm * 32 + mi * 16 + half * 8 + r;
      float inv = gamma[o] * rsqrtf(var[o] + EPSBN);
      float sh  = beta[o] + (bias[o] - mean[o]) * inv;
      #pragma unroll
      for (int ni = 0; ni < 4; ni++) {
        int n = n0 + wn * 64 + ni * 16 + l16;
        float yv = acc[mi][ni][r] * inv + sh;
        out[((size_t)b * C_OUT + o) * NPIX + n] = fmaxf(yv, 0.f);
      }
    }
  }
}

// ---------------------------------------------------------------------------
extern "C" void kernel_launch(void* const* d_in, const int* in_sizes, int n_in,
                              void* d_out, int out_size, void* d_ws, size_t ws_size,
                              hipStream_t stream) {
  const float* x     = (const float*)d_in[0];
  const float* w_off = (const float*)d_in[1];
  const float* b_off = (const float*)d_in[2];
  const float* w     = (const float*)d_in[3];
  const float* bias  = (const float*)d_in[4];
  const float* gamma = (const float*)d_in[5];
  const float* beta  = (const float*)d_in[6];
  const float* rmean = (const float*)d_in[7];
  const float* rvar  = (const float*)d_in[8];

  char* ws = (char*)d_ws;
  // workspace layout (all 16B aligned):
  //   w_bf   : 256*2304 bf16    = 1,179,648 B
  //   wof_bf : 64*2304  bf16    =   294,912 B
  //   om     : 4*54*4096 f32    = 3,538,944 B
  //   valt   : 4*4096*2304 bf16 = 75,497,472 B  (transposed [b][n][k])
  bf16*  w_bf   = (bf16*)(ws);
  bf16*  wof_bf = (bf16*)(ws + 1179648);
  float* om     = (float*)(ws + 1179648 + 294912);
  bf16*  valt   = (bf16*)(ws + 1179648 + 294912 + 3538944);

  prep_weights<<<2880, 256, 0, stream>>>(w, w_off, w_bf, wof_bf);
  offset_conv_wmma<<<dim3(16, NB), 256, 0, stream>>>(x, wof_bf, b_off, om);
  deform_sample<<<dim3(16, DG * K2, NB), 256, 0, stream>>>(x, om, valt);
  main_gemm_wmma<<<dim3(32, 2, NB), 256, 0, stream>>>(
      w_bf, valt, bias, gamma, beta, rmean, rvar, (float*)d_out);
}
